// LlamaAttention_90512140796683
// MI455X (gfx1250) — compile-verified, matrix operand order checked by reading, not run on hardware
//
#include <hip/hip_runtime.h>

// ---------------------------------------------------------------------------
// Llama attention layer for MI455X (gfx1250, wave32, WMMA bf16 + TDM).
// Compute-bound (~0.5 TFLOP vs ~290MB traffic) -> all matmuls routed through
// v_wmma_f32_16x16x32_bf16; tile staging via tensor_load_to_lds (TENSORcnt),
// double-buffered so the DMA of tile k+1 overlaps WMMA on tile k.
// ---------------------------------------------------------------------------

typedef __attribute__((ext_vector_type(16))) __bf16 v16bf;
typedef __attribute__((ext_vector_type(8)))  float  v8f;
typedef unsigned int v4u __attribute__((ext_vector_type(4)));
typedef unsigned int v8u __attribute__((ext_vector_type(8)));
typedef __bf16 bf16;

#define HIDDEN 4096
#define NH     32
#define NKV    8
#define HD     128
#define QSZ    4096          // NH*HD
#define KVSZ   1024          // NKV*HD
#define ESZ    6144          // QSZ + 2*KVSZ (qkv row width)
#define BB     2
#define SS     2048
#define MROWS  (BB*SS)       // 4096 token rows
static constexpr float SCALE_F = 0.08838834764831845f;   // 128^-0.5

// ---------------- TDM: async 2D bf16 tile load, global -> LDS ----------------
// D# per CDNA5 ISA. 2-group form (2D tensor):
//   group0: [count=1][lds_addr][global_addr 57b][type=2]
//   group1: [data_size=2B][tensor dims (huge => no OOB clip)][tile dims][stride]
// Issued once per wave; completion via s_wait_tensorcnt.

__device__ __forceinline__ void tdm_load_2d_bf16(const bf16* gsrc, unsigned lds_off,
                                                 unsigned rows, unsigned rowlen,
                                                 unsigned strideElems) {
  unsigned long long ga = (unsigned long long)(size_t)gsrc;
  v4u g0;
  g0[0] = 1u;                                              // count=1, user D#
  g0[1] = lds_off;                                         // LDS byte address
  g0[2] = (unsigned)(ga & 0xFFFFFFFFu);                    // global_addr[31:0]
  g0[3] = (unsigned)((ga >> 32) & 0x1FFFFFFu) | (2u << 30);// addr[56:32] | type=2
  v8u g1;
  g1[0] = 0x00010000u;                  // workgroup_mask=0, data_size=1 (2B)
  g1[1] = 0xFFFF0000u;                  // tensor_dim0 = 0x7FFFFFFF (lo16)
  g1[2] = 0xFFFF7FFFu;                  // tensor_dim0 hi | tensor_dim1 lo
  g1[3] = 0x00007FFFu | (rowlen << 16); // tensor_dim1 hi | tile_dim0
  g1[4] = rows;                         // tile_dim1 | tile_dim2=0
  g1[5] = strideElems;                  // tensor_dim0_stride[31:0] (elem units)
  g1[6] = 0u;                           // stride hi | dim1_stride lo
  g1[7] = 0u;
  asm volatile("tensor_load_to_lds %0, %1" :: "s"(g0), "s"(g1) : "memory");
}

// ---------------- conversion kernels ----------------

__global__ void k_cvt_bf16(const float* __restrict__ src, bf16* __restrict__ dst, int n) {
  int i = blockIdx.x * 256 + threadIdx.x;
  if (i < n) dst[i] = (bf16)src[i];
}

// src: f32 row-major (K x N). dst: bf16 row-major (N x K)  == src transposed.
__global__ void k_tr_bf16(const float* __restrict__ src, bf16* __restrict__ dst, int K, int N) {
  long long i = (long long)blockIdx.x * 256 + threadIdx.x;
  if (i < (long long)K * N) {
    int n = (int)(i / K), k = (int)(i % K);
    dst[i] = (bf16)src[(long long)k * N + n];
  }
}

// ---------------- tiled WMMA GEMM:  C[MxN] = A[MxK] * Bt[NxK]^T ----------------
// Block: 256 threads = 8 wave32 waves, tile 128x128, K-step 32.
// Staging: wave 0 issues two TDM descriptors per K-step into the ping-pong
// LDS buffers; compute overlaps the DMA for the next tile.

template <bool OUT_BF16>
__global__ __launch_bounds__(256)
void k_gemm(const bf16* __restrict__ A, const bf16* __restrict__ Bt,
            void* __restrict__ Cv, int M, int N, int K) {
  __shared__ __align__(32) bf16 As[2][128][32];
  __shared__ __align__(32) bf16 Bs[2][128][32];
  const int tid  = threadIdx.x;
  const int lane = tid & 31, wave = tid >> 5;
  const int lm = lane & 15, lh = lane >> 4;
  const int mBlk = blockIdx.y * 128, nBlk = blockIdx.x * 128;
  const int mW = (wave & 1) * 64;   // wave M base in tile
  const int nW = (wave >> 1) * 32;  // wave N base in tile

  v8f acc[4][2];
#pragma unroll
  for (int i = 0; i < 4; i++)
#pragma unroll
    for (int j = 0; j < 2; j++) acc[i][j] = {};

  if (wave == 0) {  // prime buffer 0
    tdm_load_2d_bf16(&A [(long long)mBlk * K], (unsigned)(size_t)&As[0][0][0], 128, 32, (unsigned)K);
    tdm_load_2d_bf16(&Bt[(long long)nBlk * K], (unsigned)(size_t)&Bs[0][0][0], 128, 32, (unsigned)K);
  }

  int p = 0;
  for (int k0 = 0; k0 < K; k0 += 32) {
    if (wave == 0) __builtin_amdgcn_s_wait_tensorcnt(0);
    __syncthreads();                       // tiles in buf[p] visible to all
    if (wave == 0 && k0 + 32 < K) {        // kick DMA for next K-step
      tdm_load_2d_bf16(&A [(long long)mBlk * K + k0 + 32], (unsigned)(size_t)&As[p ^ 1][0][0], 128, 32, (unsigned)K);
      tdm_load_2d_bf16(&Bt[(long long)nBlk * K + k0 + 32], (unsigned)(size_t)&Bs[p ^ 1][0][0], 128, 32, (unsigned)K);
    }
    v16bf af[4], bfv[2];
#pragma unroll
    for (int mt = 0; mt < 4; mt++) af[mt]  = *(const v16bf*)&As[p][mW + mt * 16 + lm][lh * 16];
#pragma unroll
    for (int nt = 0; nt < 2; nt++) bfv[nt] = *(const v16bf*)&Bs[p][nW + nt * 16 + lm][lh * 16];
#pragma unroll
    for (int mt = 0; mt < 4; mt++)
#pragma unroll
      for (int nt = 0; nt < 2; nt++)
        acc[mt][nt] = __builtin_amdgcn_wmma_f32_16x16x32_bf16(
            false, af[mt], false, bfv[nt], (short)0, acc[mt][nt], false, false);
    p ^= 1;
  }
  // C/D layout: vgpr r, lane L -> M = r + (L>=16)*8, N = L&15
#pragma unroll
  for (int mt = 0; mt < 4; mt++)
#pragma unroll
    for (int nt = 0; nt < 2; nt++)
#pragma unroll
      for (int r = 0; r < 8; r++) {
        int m = mBlk + mW + mt * 16 + r + lh * 8;
        int n = nBlk + nW + nt * 16 + lm;
        float v = acc[mt][nt][r];
        if (OUT_BF16) ((bf16*)Cv)[(long long)m * N + n] = (bf16)v;
        else          ((float*)Cv)[(long long)m * N + n] = v;
      }
}

// ---------------- RoPE (in place on qkv bf16) ----------------

__global__ void k_rope(const int* __restrict__ pos, bf16* __restrict__ qkv) {
  int row = blockIdx.x;                 // b*S + s
  int gh  = blockIdx.y;                 // 0..39 (32 q heads then 8 kv heads)
  int d   = threadIdx.x;                // 0..63
  int col0 = (gh < NH) ? gh * HD : QSZ + (gh - NH) * HD;
  float p   = (float)pos[row];
  float inv = __expf(-9.2103403719761836f * (float)d * (1.0f / 64.0f)); // 10000^(-d/64)
  float fr  = p * inv;
  float c = __cosf(fr), s = __sinf(fr);
  long long base = (long long)row * ESZ + col0;
  float x1 = (float)qkv[base + d];
  float x2 = (float)qkv[base + d + 64];
  qkv[base + d]      = (bf16)(x1 * c - x2 * s);
  qkv[base + d + 64] = (bf16)(x2 * c + x1 * s);
}

// ---------------- flash attention ----------------
// Block: 128 threads = 4 waves; block owns 64 q rows of one (b, head);
// each wave owns 16 q rows. Per 32-key chunk: wave0 issues a TDM for the
// K tile (overlapping the manual V transpose), 8 WMMAs for Q*K^T, online
// softmax, 8 WMMAs for P*V.

__global__ __launch_bounds__(128)
void k_attn(const bf16* __restrict__ qkv, bf16* __restrict__ attn) {
  __shared__ __align__(32) bf16 Ks[32][128];   // K chunk, [key][dim] (TDM)
  __shared__ __align__(32) bf16 Vt[128][32];   // V chunk transposed, [dim][key]
  __shared__ __align__(32) bf16 Pb[4][16][32]; // per-wave P relayout buffer

  const int tid = threadIdx.x, lane = tid & 31, wave = tid >> 5;
  const int lm = lane & 15, lh = lane >> 4;
  const int qBlk = blockIdx.x * 64;
  const int h = blockIdx.y, b = blockIdx.z;
  const int kvh = h >> 2;                       // GQA: rep = 32/8 = 4
  const int qRow0 = qBlk + wave * 16;
  const long long qkvBase = (long long)b * SS * ESZ;

  // Q fragments (A-operand): lane holds row lm, dims lh*16..+15 contiguous
  v16bf qf[4];
  {
    long long qoff = qkvBase + (long long)(qRow0 + lm) * ESZ + h * HD + lh * 16;
#pragma unroll
    for (int f = 0; f < 4; f++) qf[f] = *(const v16bf*)&qkv[qoff + f * 32];
  }

  v8f o[8];
#pragma unroll
  for (int i = 0; i < 8; i++) o[i] = {};
  float rm[8], rs[8];
#pragma unroll
  for (int r = 0; r < 8; r++) { rm[r] = -1e30f; rs[r] = 0.f; }

  const int kLimit = qBlk + 64;                 // causal: keys <= max q row in block
  for (int kb = 0; kb < kLimit; kb += 32) {
    // K tile via TDM (32 rows x 128 elems, row stride ESZ)
    if (wave == 0)
      tdm_load_2d_bf16(&qkv[qkvBase + (long long)kb * ESZ + QSZ + kvh * HD],
                       (unsigned)(size_t)&Ks[0][0], 32, 128, (unsigned)ESZ);
    // V transpose: each thread packs 4 keys x 8 dims -> 8x ds_store_b64
    {
      int kq = tid >> 4, c8 = (tid & 15) * 8;   // key quad 0..7, dim base 0..120
      long long vb = qkvBase + QSZ + KVSZ + kvh * HD + c8;
      union { uint4 u; unsigned short h8[8]; } cv[4];
#pragma unroll
      for (int j = 0; j < 4; j++)
        cv[j].u = *(const uint4*)&qkv[vb + (long long)(kb + kq * 4 + j) * ESZ];
#pragma unroll
      for (int e = 0; e < 8; e++) {
        unsigned lo = (unsigned)cv[0].h8[e] | ((unsigned)cv[1].h8[e] << 16);
        unsigned hi = (unsigned)cv[2].h8[e] | ((unsigned)cv[3].h8[e] << 16);
        *(unsigned long long*)&Vt[c8 + e][kq * 4] =
            (unsigned long long)lo | ((unsigned long long)hi << 32);
      }
    }
    if (wave == 0) __builtin_amdgcn_s_wait_tensorcnt(0);
    __syncthreads();

    if (kb < qRow0 + 16) {                      // wave-uniform causal skip
      v8f sa[2]; sa[0] = {}; sa[1] = {};
#pragma unroll
      for (int f = 0; f < 4; f++) {
        v16bf kf0 = *(const v16bf*)&Ks[lm]     [f * 32 + lh * 16];
        v16bf kf1 = *(const v16bf*)&Ks[16 + lm][f * 32 + lh * 16];
        sa[0] = __builtin_amdgcn_wmma_f32_16x16x32_bf16(false, qf[f], false, kf0, (short)0, sa[0], false, false);
        sa[1] = __builtin_amdgcn_wmma_f32_16x16x32_bf16(false, qf[f], false, kf1, (short)0, sa[1], false, false);
      }
      // online softmax; row stats reduced across the 16 lanes of each half-wave
#pragma unroll
      for (int r = 0; r < 8; r++) {
        int qi = qRow0 + r + lh * 8;
        float s0 = (kb + lm      <= qi) ? sa[0][r] * SCALE_F : -1e30f;
        float s1 = (kb + 16 + lm <= qi) ? sa[1][r] * SCALE_F : -1e30f;
        float mx = fmaxf(s0, s1);
#pragma unroll
        for (int msk = 1; msk < 16; msk <<= 1) mx = fmaxf(mx, __shfl_xor(mx, msk, 32));
        float mnew  = fmaxf(rm[r], mx);
        float alpha = __expf(rm[r] - mnew);
        rm[r] = mnew;
        float e0 = __expf(s0 - mnew), e1 = __expf(s1 - mnew);
        float sum = e0 + e1;
#pragma unroll
        for (int msk = 1; msk < 16; msk <<= 1) sum += __shfl_xor(sum, msk, 32);
        rs[r] = rs[r] * alpha + sum;
#pragma unroll
        for (int dt = 0; dt < 8; dt++) o[dt][r] *= alpha;
        // stash P (C-layout -> LDS) for A-fragment relayout
        Pb[wave][r + lh * 8][lm]      = (bf16)e0;
        Pb[wave][r + lh * 8][16 + lm] = (bf16)e1;
      }
      asm volatile("s_wait_dscnt 0" ::: "memory");  // intra-wave LDS RAW fence
      v16bf pf = *(const v16bf*)&Pb[wave][lm][lh * 16];
#pragma unroll
      for (int dt = 0; dt < 8; dt++) {
        v16bf vf = *(const v16bf*)&Vt[dt * 16 + lm][lh * 16];
        o[dt] = __builtin_amdgcn_wmma_f32_16x16x32_bf16(false, pf, false, vf, (short)0, o[dt], false, false);
      }
    }
    __syncthreads();                            // before restaging Ks/Vt
  }

  // normalize and store bf16 attention output (B,S,QSZ)
#pragma unroll
  for (int r = 0; r < 8; r++) {
    float inv = 1.0f / rs[r];
    int qi = qRow0 + r + lh * 8;
    long long obase = ((long long)b * SS + qi) * QSZ + h * HD;
#pragma unroll
    for (int dt = 0; dt < 8; dt++)
      attn[obase + dt * 16 + lm] = (bf16)(o[dt][r] * inv);
  }
}

// ---------------- launcher ----------------
// Workspace layout (128 MiB total, aliased across phases):
//   [0,   32Mi) : X bf16            -> reused for W_o^T bf16 after QKV GEMM
//   [32,  80Mi) : W_qkv^T bf16      -> reused for attn bf16 after QKV GEMM
//   [80, 128Mi) : qkv bf16 (B,S,6144)

extern "C" void kernel_launch(void* const* d_in, const int* in_sizes, int n_in,
                              void* d_out, int out_size, void* d_ws, size_t ws_size,
                              hipStream_t stream) {
  const int*   positions = (const int*)  d_in[0];
  const float* hidden    = (const float*)d_in[1];
  const float* W_qkv     = (const float*)d_in[2];
  const float* W_o       = (const float*)d_in[3];
  (void)in_sizes; (void)n_in; (void)out_size; (void)ws_size;

  char* ws = (char*)d_ws;
  bf16* Xb   = (bf16*)(ws);
  bf16* Wt   = (bf16*)(ws + (size_t)32 * 1024 * 1024);
  bf16* qkvb = (bf16*)(ws + (size_t)80 * 1024 * 1024);
  bf16* Wot  = Xb;   // alias: X dead after QKV GEMM
  bf16* attn = Wt;   // alias: W_qkv^T dead after QKV GEMM
  float* out = (float*)d_out;

  // 1) X -> bf16
  { int n = MROWS * HIDDEN;
    k_cvt_bf16<<<n / 256, 256, 0, stream>>>(hidden, Xb, n); }
  // 2) W_qkv^T -> bf16  [ESZ x HIDDEN]
  { long long n = (long long)ESZ * HIDDEN;
    k_tr_bf16<<<(int)(n / 256), 256, 0, stream>>>(W_qkv, Wt, HIDDEN, ESZ); }
  // 3) qkv = X * W_qkv   (bf16 out)
  k_gemm<true ><<<dim3(ESZ / 128, MROWS / 128), 256, 0, stream>>>(Xb, Wt, qkvb, MROWS, ESZ, HIDDEN);
  // 4) RoPE on q/k slices
  k_rope<<<dim3(MROWS, NH + NKV), 64, 0, stream>>>(positions, qkvb);
  // 5) W_o^T -> bf16  [HIDDEN x QSZ]  (over dead X region)
  { long long n = (long long)HIDDEN * QSZ;
    k_tr_bf16<<<(int)(n / 256), 256, 0, stream>>>(W_o, Wot, QSZ, HIDDEN); }
  // 6) flash attention -> attn bf16 (over dead W_qkv^T region)
  k_attn<<<dim3(SS / 64, NH, BB), 128, 0, stream>>>(qkvb, attn);
  // 7) out = attn * W_o  (f32 out)
  k_gemm<false><<<dim3(HIDDEN / 128, MROWS / 128), 256, 0, stream>>>(attn, Wot, out, MROWS, HIDDEN, HIDDEN);
}
